// TSGUARD_11321533792838
// MI455X (gfx1250) — matrix operand order repaired and checked by reading, not compiled, not run on hardware
//
#include <hip/hip_runtime.h>
#include <hip/hip_bf16.h>
#include <stdint.h>

typedef __attribute__((ext_vector_type(16))) _Float16 v16h;
typedef __attribute__((ext_vector_type(8)))  _Float16 v8h;
typedef __attribute__((ext_vector_type(8)))  float    v8f;

// ---------------------------------------------------------------------------
// Degree / normalization kernels
// ---------------------------------------------------------------------------
__global__ void k_deg_init(float* __restrict__ deg, int N) {
  int i = blockIdx.x * blockDim.x + threadIdx.x;
  if (i < N) deg[i] = 1.0f;  // self-loop contributes 1 to every node's degree
}

__global__ void k_deg_accum(const long long* __restrict__ dst, float* __restrict__ deg,
                            long long E) {
  long long e = (long long)blockIdx.x * blockDim.x + threadIdx.x;
  if (e < E) atomicAdd(&deg[(int)dst[e]], 1.0f);
}

__global__ void k_dinv(const float* __restrict__ deg, float* __restrict__ dinv, int N) {
  int i = blockIdx.x * blockDim.x + threadIdx.x;
  if (i < N) {
    float d = deg[i];
    dinv[i] = (d > 0.0f) ? rsqrtf(d) : 0.0f;
  }
}

// ---------------------------------------------------------------------------
// f32 -> f16 conversion (A-matrix activations, row-major kept)
// ---------------------------------------------------------------------------
__global__ void k_cvt_f16(const float* __restrict__ in, _Float16* __restrict__ out,
                          long long n) {
  long long i = (long long)blockIdx.x * blockDim.x + threadIdx.x;
  if (i < n) out[i] = (_Float16)in[i];
}

// ---------------------------------------------------------------------------
// Pack W[K,N] f32 (row-major) into the WMMA B-operand per-lane layout, f16:
//   Bp[(((kb*NT + nt)*32 + lane)*16 + j] = W[(kb*32 + (lane>>4)*8 + (j&7) + (j>>3)*16)*N + nt*16 + (lane&15)]
// so in the GEMM each lane loads its 16 B-halfs as one contiguous 32-byte block.
// ---------------------------------------------------------------------------
template <int N>
__global__ void k_pack_B(const float* __restrict__ W, _Float16* __restrict__ Bp, int K) {
  int idx = blockIdx.x * blockDim.x + threadIdx.x;
  if (idx >= K * N) return;
  constexpr int NT = N / 16;
  int j    = idx & 15;
  int lane = (idx >> 4) & 31;
  int t    = idx >> 9;            // kb*NT + nt
  int kb   = t / NT;
  int nt   = t - kb * NT;
  int col  = nt * 16 + (lane & 15);
  int krow = kb * 32 + (lane >> 4) * 8 + (j & 7) + (j >> 3) * 16;
  Bp[idx] = (_Float16)W[krow * N + col];
}

// ---------------------------------------------------------------------------
// WMMA GEMM: C[M,N] = A[M,K] * Bp, one wave32 per 16-row strip of C.
// A: f16 row-major. Bp: pre-packed (see k_pack_B). C: f32 row-major.
// Fully unrolled: (K/32) k-steps x (N/16) n-tiles independent WMMAs,
// A fragment loaded once per k-step and reused across all n-tiles.
//
// A operand (ISA 7.12.2, 16-bit A 16x32): lane (l&15)=row, hi=(l>>4):
//   elems 0..7 = K[8*hi..8*hi+7], elems 8..15 = K[16+8*hi..16+8*hi+7]
// D (16x16 f32): vgpr r, lane l -> row m0 + 8*(l>>4) + r, col n0 + (l&15)
// ---------------------------------------------------------------------------
template <int K, int N>
__global__ void k_gemm_wmma_f16(const _Float16* __restrict__ A,
                                const _Float16* __restrict__ Bp,
                                float* __restrict__ C, int M) {
  constexpr int NT = N / 16;
  constexpr int KB = K / 32;
  int wave = (int)(((long long)blockIdx.x * blockDim.x + threadIdx.x) >> 5);
  int lane = threadIdx.x & 31;
  if (wave >= (M >> 4)) return;      // uniform per-wave: EXEC stays all-1s
  int m0 = wave << 4;
  int hl = lane & 15;
  int hi = lane >> 4;

  v8f acc[NT];
#pragma unroll
  for (int nt = 0; nt < NT; ++nt) acc[nt] = (v8f){};

  const _Float16* arow = A + (size_t)(m0 + hl) * K;

#pragma unroll
  for (int kb = 0; kb < KB; ++kb) {
    int ka = kb * 32 + hi * 8;
    v8h a0 = *(const v8h*)(arow + ka);        // K[ka .. ka+7]
    v8h a1 = *(const v8h*)(arow + ka + 16);   // K[ka+16 .. ka+23]
    v16h a = __builtin_shufflevector(a0, a1, 0, 1, 2, 3, 4, 5, 6, 7,
                                     8, 9, 10, 11, 12, 13, 14, 15);
#pragma unroll
    for (int nt = 0; nt < NT; ++nt) {
      v16h b = *(const v16h*)(Bp + ((size_t)(kb * NT + nt) * 32 + lane) * 16);
      acc[nt] = __builtin_amdgcn_wmma_f32_16x16x32_f16(
          /*neg_a=*/false, a, /*neg_b=*/false, b,
          /*c_mod=*/(short)0, acc[nt], /*reuse_a=*/false, /*reuse_b=*/false);
    }
  }

  float* crow = C + (size_t)(m0 + 8 * hi) * N + hl;
#pragma unroll
  for (int nt = 0; nt < NT; ++nt)
#pragma unroll
    for (int r = 0; r < 8; ++r)
      crow[(size_t)r * N + nt * 16] = acc[nt][r];
}

// ---------------------------------------------------------------------------
// Aggregation: Out[i,:] = dinv[i]^2 * T[i,:] + bias   (self-loop + bias init)
// then per edge: Out[dst,:] += dinv[src]*dinv[dst] * T[src,:]  (atomics)
// ---------------------------------------------------------------------------
template <int F>
__global__ void k_self_bias_init(const float* __restrict__ T,
                                 const float* __restrict__ dinv,
                                 const float* __restrict__ bias,
                                 float* __restrict__ Out, int N) {
  long long i = (long long)blockIdx.x * blockDim.x + threadIdx.x;
  if (i >= (long long)N * F) return;
  int node = (int)(i >> __builtin_ctz(F));
  int f    = (int)(i & (F - 1));
  float di = dinv[node];
  Out[i] = T[i] * di * di + bias[f];
}

template <int F>
__global__ void k_edge_agg(const long long* __restrict__ src,
                           const long long* __restrict__ dst,
                           const float* __restrict__ dinv,
                           const float* __restrict__ T,
                           float* __restrict__ Out, long long E) {
  long long idx = (long long)blockIdx.x * blockDim.x + threadIdx.x;
  constexpr int CH = F / 4;                  // float4 chunks per edge
  long long e = idx / CH;
  if (e >= E) return;
  int c = (int)(idx - e * CH) << 2;

  int s = (int)src[e];
  int d = (int)dst[e];
  float nrm = dinv[s] * dinv[d];

  const float4 v = *(const float4*)(T + (long long)s * F + c);
  float* o = Out + (long long)d * F + c;
  atomicAdd(o + 0, v.x * nrm);
  atomicAdd(o + 1, v.y * nrm);
  atomicAdd(o + 2, v.z * nrm);
  atomicAdd(o + 3, v.w * nrm);
}

// relu + f16 convert (feeds second WMMA GEMM)
__global__ void k_relu_cvt_f16(const float* __restrict__ in, _Float16* __restrict__ out,
                               long long n) {
  long long i = (long long)blockIdx.x * blockDim.x + threadIdx.x;
  if (i < n) out[i] = (_Float16)fmaxf(in[i], 0.0f);
}

// ---------------------------------------------------------------------------
// Launcher
// ---------------------------------------------------------------------------
extern "C" void kernel_launch(void* const* d_in, const int* in_sizes, int n_in,
                              void* d_out, int out_size, void* d_ws, size_t ws_size,
                              hipStream_t stream) {
  const float*     x  = (const float*)d_in[0];
  const long long* ei = (const long long*)d_in[1];   // int64 [2,E]
  const float*     W1 = (const float*)d_in[2];
  const float*     b1 = (const float*)d_in[3];
  const float*     W2 = (const float*)d_in[4];
  const float*     b2 = (const float*)d_in[5];

  constexpr int Fin = 128, Fh = 128, Fo = 64;
  const int N = in_sizes[0] / Fin;
  const long long E = (long long)in_sizes[1] / 2;
  const long long* srcI = ei;
  const long long* dstI = ei + E;

  // Workspace carve-out (offsets are 64B-aligned for these sizes)
  char* ws = (char*)d_ws;
  float*    deg  = (float*)ws;                         // N
  float*    dinv = deg + N;                            // N
  _Float16* W1p  = (_Float16*)(dinv + N);              // Fin*Fh  (packed B)
  _Float16* W2p  = W1p + (size_t)Fin * Fh;             // Fh*Fo   (packed B)
  _Float16* Xh   = W2p + (size_t)Fh * Fo;              // N*Fin   (reused as H1h)
  float*    T1   = (float*)(Xh + (size_t)N * Fin);     // N*Fh    (reused as T2)
  float*    A1   = T1 + (size_t)N * Fh;                // N*Fh
  _Float16* H1h  = Xh;                                 // reuse: Xh dead after gemm1
  float*    T2   = T1;                                 // reuse: T1 dead after relu
  float*    outF = (float*)d_out;                      // N*Fo

  const int B256 = 256;
  auto blocks = [](long long n, int b) { return (unsigned)((n + b - 1) / b); };

  // --- degrees & normalization ---
  k_deg_init <<<blocks(N, B256), B256, 0, stream>>>(deg, N);
  k_deg_accum<<<blocks(E, B256), B256, 0, stream>>>(dstI, deg, E);
  k_dinv     <<<blocks(N, B256), B256, 0, stream>>>(deg, dinv, N);

  // --- weight packing + activation conversion ---
  k_pack_B<Fh><<<blocks((long long)Fin * Fh, B256), B256, 0, stream>>>(W1, W1p, Fin);
  k_pack_B<Fo><<<blocks((long long)Fh * Fo,  B256), B256, 0, stream>>>(W2, W2p, Fh);
  k_cvt_f16<<<blocks((long long)N * Fin, B256), B256, 0, stream>>>(x, Xh, (long long)N * Fin);

  // --- layer 1: T1 = Xh @ W1 (WMMA, one wave per 16-row strip) ---
  k_gemm_wmma_f16<Fin, Fh><<<blocks((long long)(N / 16) * 32, B256), B256, 0, stream>>>(
      Xh, W1p, T1, N);
  // A1 = dinv^2*T1 + b1 ; edge scatter ; relu -> H1h
  k_self_bias_init<Fh><<<blocks((long long)N * Fh, B256), B256, 0, stream>>>(T1, dinv, b1, A1, N);
  k_edge_agg<Fh><<<blocks(E * (Fh / 4), B256), B256, 0, stream>>>(srcI, dstI, dinv, T1, A1, E);
  k_relu_cvt_f16<<<blocks((long long)N * Fh, B256), B256, 0, stream>>>(A1, H1h, (long long)N * Fh);

  // --- layer 2: T2 = H1h @ W2 (WMMA) ---
  k_gemm_wmma_f16<Fh, Fo><<<blocks((long long)(N / 16) * 32, B256), B256, 0, stream>>>(
      H1h, W2p, T2, N);
  // out = dinv^2*T2 + b2 ; edge scatter into d_out
  k_self_bias_init<Fo><<<blocks((long long)N * Fo, B256), B256, 0, stream>>>(T2, dinv, b2, outF, N);
  k_edge_agg<Fo><<<blocks(E * (Fo / 4), B256), B256, 0, stream>>>(srcI, dstI, dinv, T2, outF, E);
}
